// RNNPOSTagger_52578989637833
// MI455X (gfx1250) — matrix operand order repaired and checked by reading, not compiled, not run on hardware
//
#include <hip/hip_runtime.h>
#include <hip/hip_bf16.h>

typedef __attribute__((ext_vector_type(16))) __bf16        v16bf;
typedef __attribute__((ext_vector_type(8)))  float         v8f;
typedef __attribute__((ext_vector_type(8)))  unsigned int  v8u;

#define L_SEQ 512
#define BATCH 64
#define EDIM  256
#define HDIM  256
#define TDIM  50
#define ROWS  (L_SEQ * BATCH)   // 32768
#define WPITCH 264              // padded LDS pitch (ushorts) to avoid bank conflicts

#define WMMA_BF16(a, b, c) \
  __builtin_amdgcn_wmma_f32_16x16x32_bf16(false, (a), false, (b), (short)0, (c), false, false)

// ---- helpers -------------------------------------------------------------

__device__ __forceinline__ unsigned short f2bf(float f) {
  return __builtin_bit_cast(unsigned short, (__bf16)f);   // native cvt, RTNE
}

__device__ __forceinline__ unsigned pack_bf(float lo, float hi) {
  return (unsigned)f2bf(lo) | ((unsigned)f2bf(hi) << 16);
}

// B fragment: elems 0..7 at p[0..7], elems 8..15 at p[16..23] (column n of B = row n of row-major W)
__device__ __forceinline__ v16bf load_bfrag(const unsigned short* p) {
  uint4 lo = *(const uint4*)p;
  uint4 hi = *(const uint4*)(p + 16);
  v8u r;
  r[0] = lo.x; r[1] = lo.y; r[2] = lo.z; r[3] = lo.w;
  r[4] = hi.x; r[5] = hi.y; r[6] = hi.z; r[7] = hi.w;
  return __builtin_bit_cast(v16bf, r);
}

// A fragment from bf16 source: elems 0..7 at p[0..7], elems 8..15 at p[16..23]
__device__ __forceinline__ v16bf load_afrag_bf(const unsigned short* p) {
  uint4 lo = *(const uint4*)p;
  uint4 hi = *(const uint4*)(p + 16);
  v8u r;
  r[0] = lo.x; r[1] = lo.y; r[2] = lo.z; r[3] = lo.w;
  r[4] = hi.x; r[5] = hi.y; r[6] = hi.z; r[7] = hi.w;
  return __builtin_bit_cast(v16bf, r);
}

// A fragment from fp32 source: convert 8 floats at p[0..7] and 8 at p[16..23]
__device__ __forceinline__ v16bf afrag_from_f32(const float* p) {
  float4 x0 = ((const float4*)p)[0];
  float4 x1 = ((const float4*)p)[1];
  float4 x2 = ((const float4*)(p + 16))[0];
  float4 x3 = ((const float4*)(p + 16))[1];
  v8u r;
  r[0] = pack_bf(x0.x, x0.y); r[1] = pack_bf(x0.z, x0.w);
  r[2] = pack_bf(x1.x, x1.y); r[3] = pack_bf(x1.z, x1.w);
  r[4] = pack_bf(x2.x, x2.y); r[5] = pack_bf(x2.z, x2.w);
  r[6] = pack_bf(x3.x, x3.y); r[7] = pack_bf(x3.z, x3.w);
  return __builtin_bit_cast(v16bf, r);
}

// ---- kernel 1: weight prep (fp32 -> bf16, split Wx/Wh, pad W_out) --------

__global__ void prep_kernel(const float* __restrict__ Wf, const float* __restrict__ Wb,
                            const float* __restrict__ Wout,
                            unsigned short* __restrict__ wxf, unsigned short* __restrict__ whf,
                            unsigned short* __restrict__ wxb, unsigned short* __restrict__ whb,
                            unsigned short* __restrict__ wo) {
  int idx = blockIdx.x * blockDim.x + threadIdx.x;
  if (idx < HDIM * EDIM) {            // 65536
    int n = idx >> 8, k = idx & 255;
    wxf[idx] = f2bf(Wf[n * 512 + k]);
    whf[idx] = f2bf(Wf[n * 512 + 256 + k]);
    wxb[idx] = f2bf(Wb[n * 512 + k]);
    whb[idx] = f2bf(Wb[n * 512 + 256 + k]);
  }
  if (idx < 64 * 512) {               // 32768, padded to 64 output rows
    int n = idx >> 9, k = idx & 511;
    wo[idx] = (n < TDIM) ? f2bf(Wout[n * 512 + k]) : (unsigned short)0;
  }
}

// ---- kernel 2: fused embedding gather + input projection -----------------
// Xproj is stored FRAGMENT-MAJOR: [global_mtile][ntile][lane][8] so that the
// scan kernel's accumulator seed is two b128 loads per tile.

__global__ void __launch_bounds__(256)
xproj_kernel(const int* __restrict__ text, const float* __restrict__ emb,
             const unsigned short* __restrict__ wxf, const unsigned short* __restrict__ wxb,
             const float* __restrict__ bf_, const float* __restrict__ bb_,
             float* __restrict__ xproj_f, float* __restrict__ xproj_b) {
  int wave = (blockIdx.x * blockDim.x + threadIdx.x) >> 5;
  int lane = threadIdx.x & 31;
  int dir   = wave >> 15;             // 32768 waves per direction
  int rem   = wave & 32767;
  int mtile = rem >> 4;               // 0..2047
  int ntile = rem & 15;               // 0..15

  int mr = lane & 15;
  int hi = lane >> 4;
  int row = mtile * 16 + mr;          // global row = s*64 + b
  int s = row >> 6, b = row & 63;
  int src_t = dir ? (L_SEQ - 1 - s) : s;
  int token = text[src_t * BATCH + b];

  const unsigned short* W    = dir ? wxb : wxf;
  const float*          bias = dir ? bb_ : bf_;
  float*                outp = dir ? xproj_b : xproj_f;

  const float* erow = emb + (size_t)token * EDIM;
  int n = ntile * 16 + mr;
  const unsigned short* wrow = W + n * EDIM;

  v8f acc = {};
  for (int kk = 0; kk < 8; ++kk) {
    v16bf a  = afrag_from_f32(erow + kk * 32 + hi * 8);
    v16bf bm = load_bfrag(wrow + kk * 32 + hi * 8);
    acc = WMMA_BF16(a, bm, acc);
  }

  float bv = bias[n];
  float* op = outp + ((size_t)(mtile * 16 + ntile) * 32 + lane) * 8;
  float4 o0 = make_float4(acc[0] + bv, acc[1] + bv, acc[2] + bv, acc[3] + bv);
  float4 o1 = make_float4(acc[4] + bv, acc[5] + bv, acc[6] + bv, acc[7] + bv);
  ((float4*)op)[0] = o0;
  ((float4*)op)[1] = o1;
}

// ---- kernel 3: recurrent scan, one persistent workgroup per direction ----
// Wh (128KB) + double-buffered bf16 h in LDS (~203KB of the WGP's 320KB).
// Double buffering -> single __syncthreads per step on the 512-step
// sequential critical path.

__global__ void __launch_bounds__(512)
scan_kernel(const float* __restrict__ xproj_f, const float* __restrict__ xproj_b,
            const unsigned short* __restrict__ whf, const unsigned short* __restrict__ whb,
            float* __restrict__ hidden) {
  int dir = blockIdx.x;
  const float*          Xp = dir ? xproj_b : xproj_f;
  const unsigned short* Wh = dir ? whb : whf;

  __shared__ unsigned short whl[HDIM * WPITCH];   // Wh, padded pitch
  __shared__ unsigned short hb0[64 * WPITCH];     // h ping
  __shared__ unsigned short hb1[64 * WPITCH];     // h pong

  // stage Wh into LDS (row-padded), zero initial h buffer
  for (int idx = threadIdx.x; idx < HDIM * 16; idx += blockDim.x) {
    int r = idx >> 4, seg = idx & 15;             // 16 uint4 (128 elems) per row
    ((uint4*)whl)[r * (WPITCH / 8) + seg] = ((const uint4*)Wh)[r * 16 + seg];
  }
  for (int i = threadIdx.x; i < 64 * WPITCH; i += blockDim.x) hb0[i] = 0;
  __syncthreads();

  int wave = threadIdx.x >> 5;      // 16 waves
  int lane = threadIdx.x & 31;
  int hi = lane >> 4;
  int mr = lane & 15;
  int mtile = wave & 3;             // 64 rows / 16
  int ngrp  = wave >> 2;            // 4 n-tiles per wave

  int hrow_off = (mtile * 16 + mr) * WPITCH;
  const unsigned short* wrow[4];
  int ncol[4];
  for (int j = 0; j < 4; ++j) {
    ncol[j] = (ngrp * 4 + j) * 16 + mr;
    wrow[j] = whl + ncol[j] * WPITCH;
  }

  for (int s = 0; s < L_SEQ; ++s) {
    const unsigned short* hread = (s & 1) ? hb1 : hb0;  // read buffer
    unsigned short*       hwrit = (s & 1) ? hb0 : hb1;  // write buffer (disjoint)

    v8f acc[4];
    // seed accumulators with Xproj (fragment-major: 2 x b128 per tile)
    for (int j = 0; j < 4; ++j) {
      const float* xp = Xp + (((size_t)(s * 4 + mtile) * 16 + (ngrp * 4 + j)) * 32 + lane) * 8;
      float4 a0 = ((const float4*)xp)[0];
      float4 a1 = ((const float4*)xp)[1];
      v8f a;
      a[0] = a0.x; a[1] = a0.y; a[2] = a0.z; a[3] = a0.w;
      a[4] = a1.x; a[5] = a1.y; a[6] = a1.z; a[7] = a1.w;
      acc[j] = a;
    }
    // h @ Wh.T via WMMA; A-frag shared across the wave's 4 n-tiles; all LDS
    const unsigned short* hrow = hread + hrow_off;
    for (int kk = 0; kk < 8; ++kk) {
      v16bf afrag = load_afrag_bf(hrow + kk * 32 + hi * 8);
      for (int j = 0; j < 4; ++j) {
        v16bf bfrag = load_bfrag(wrow[j] + kk * 32 + hi * 8);
        acc[j] = WMMA_BF16(afrag, bfrag, acc[j]);
      }
    }
    // sigmoid + write to the OTHER h buffer and to global hidden
    for (int j = 0; j < 4; ++j) {
      for (int i = 0; i < 8; ++i) {
        float h = 1.0f / (1.0f + __expf(-acc[j][i]));
        int r = mtile * 16 + hi * 8 + i;
        hwrit[r * WPITCH + ncol[j]] = f2bf(h);
        hidden[(size_t)(s * 64 + r) * (2 * HDIM) + dir * HDIM + ncol[j]] = h;
      }
    }
    __syncthreads();   // new h visible before next step reads it
  }
}

// ---- kernel 4: output projection  out = hidden @ Wout.T + b_out ----------

__global__ void __launch_bounds__(256)
outproj_kernel(const float* __restrict__ hidden, const unsigned short* __restrict__ wo,
               const float* __restrict__ bout, float* __restrict__ outp) {
  int wave = (blockIdx.x * blockDim.x + threadIdx.x) >> 5;
  int lane = threadIdx.x & 31;
  int hi = lane >> 4, mr = lane & 15;
  int mtile = wave >> 2;            // 0..2047
  int ntile = wave & 3;             // 0..3 (64 padded output cols)

  const float* hrow = hidden + (size_t)(mtile * 16 + mr) * 512;
  int n = ntile * 16 + mr;
  const unsigned short* wrow = wo + n * 512;

  v8f acc = {};
  for (int kk = 0; kk < 16; ++kk) {
    v16bf a  = afrag_from_f32(hrow + kk * 32 + hi * 8);
    v16bf bm = load_bfrag(wrow + kk * 32 + hi * 8);
    acc = WMMA_BF16(a, bm, acc);
  }

  if (n < TDIM) {
    float bv = bout[n];
    for (int i = 0; i < 8; ++i) {
      int m = mtile * 16 + hi * 8 + i;
      outp[(size_t)m * TDIM + n] = acc[i] + bv;
    }
  }
}

// ---- launcher ------------------------------------------------------------

extern "C" void kernel_launch(void* const* d_in, const int* in_sizes, int n_in,
                              void* d_out, int out_size, void* d_ws, size_t ws_size,
                              hipStream_t stream) {
  const int*   text = (const int*)d_in[0];
  const float* emb  = (const float*)d_in[1];
  const float* Wf   = (const float*)d_in[2];
  const float* bf_  = (const float*)d_in[3];
  const float* Wb   = (const float*)d_in[4];
  const float* bb_  = (const float*)d_in[5];
  const float* Wout = (const float*)d_in[6];
  const float* bout = (const float*)d_in[7];

  float* out    = (float*)d_out;                     // [L*B, T]
  float* hidden = out + (size_t)ROWS * TDIM;         // [L*B, 2H]

  float* xproj_f = (float*)d_ws;                         // 32768*256 f32, fragment-major
  float* xproj_b = xproj_f + (size_t)ROWS * HDIM;
  unsigned short* wxf = (unsigned short*)(xproj_b + (size_t)ROWS * HDIM);
  unsigned short* whf = wxf + HDIM * EDIM;
  unsigned short* wxb = whf + HDIM * HDIM;
  unsigned short* whb = wxb + HDIM * EDIM;
  unsigned short* wo  = whb + HDIM * HDIM;               // 64*512 bf16

  prep_kernel<<<256, 256, 0, stream>>>(Wf, Wb, Wout, wxf, whf, wxb, whb, wo);
  xproj_kernel<<<8192, 256, 0, stream>>>(text, emb, wxf, wxb, bf_, bb_, xproj_f, xproj_b);
  scan_kernel<<<2, 512, 0, stream>>>(xproj_f, xproj_b, whf, whb, hidden);
  outproj_kernel<<<1024, 256, 0, stream>>>(hidden, wo, bout, out);
}
